// MultiHeadAttention_63909113365081
// MI455X (gfx1250) — matrix operand order repaired and checked by reading, not compiled, not run on hardware
//
#include <hip/hip_runtime.h>

// ---------------------------------------------------------------------------
// MHA forward on gfx1250 (MI455X): bf16 WMMA pipeline, f32 accumulate.
// ---------------------------------------------------------------------------

typedef __bf16 bf16;
typedef __attribute__((ext_vector_type(4)))  __bf16 v4bf;
typedef __attribute__((ext_vector_type(8)))  __bf16 v8bf;
typedef __attribute__((ext_vector_type(16))) __bf16 v16bf;
typedef __attribute__((ext_vector_type(8)))  float  v8f;

#define D_MODEL 1024
#define N_HEAD  16
#define D_HEAD  64
#define SEQ_L   2048
#define BATCH   4
#define M_TOT   (BATCH * SEQ_L)   // 8192 token rows

__device__ __forceinline__ bf16 to_bf16(float f) {
    unsigned u = __builtin_bit_cast(unsigned, f);
    u += 0x7FFFu + ((u >> 16) & 1u);          // round-to-nearest-even
    unsigned short h = (unsigned short)(u >> 16);
    return __builtin_bit_cast(bf16, h);
}

__device__ __forceinline__ v16bf cat8(v8bf a, v8bf b) {
    return __builtin_shufflevector(a, b, 0,1,2,3,4,5,6,7,8,9,10,11,12,13,14,15);
}

__device__ __forceinline__ v8f wmma_bf16(v16bf a, v16bf b, v8f c) {
    return __builtin_amdgcn_wmma_f32_16x16x32_bf16(
        /*neg_a=*/false, a, /*neg_b=*/false, b,
        /*c_mod=*/(short)0, c, /*reuse_a=*/false, /*reuse_b=*/false);
}

// ---- 16-lane butterfly reductions in pure VALU via v_permlane16_b32 -------
#if __has_builtin(__builtin_amdgcn_permlane16)
__device__ __forceinline__ float pl16(float x, unsigned lo, unsigned hi) {
    int xi = __builtin_bit_cast(int, x);
    int r = __builtin_amdgcn_permlane16(xi, xi, lo, hi, false, false);
    return __builtin_bit_cast(float, r);
}
__device__ __forceinline__ float rowmax16(float x) {
    x = fmaxf(x, pl16(x, 0x67452301u, 0xEFCDAB89u)); // xor 1
    x = fmaxf(x, pl16(x, 0x54761032u, 0xDCFE98BAu)); // xor 2
    x = fmaxf(x, pl16(x, 0x32107654u, 0xBA98FEDCu)); // xor 4
    x = fmaxf(x, pl16(x, 0xFEDCBA98u, 0x76543210u)); // xor 8
    return x;
}
__device__ __forceinline__ float rowsum16(float x) {
    x += pl16(x, 0x67452301u, 0xEFCDAB89u);
    x += pl16(x, 0x54761032u, 0xDCFE98BAu);
    x += pl16(x, 0x32107654u, 0xBA98FEDCu);
    x += pl16(x, 0xFEDCBA98u, 0x76543210u);
    return x;
}
#else
__device__ __forceinline__ float rowmax16(float x) {
    #pragma unroll
    for (int m = 1; m < 16; m <<= 1) x = fmaxf(x, __shfl_xor(x, m, 16));
    return x;
}
__device__ __forceinline__ float rowsum16(float x) {
    #pragma unroll
    for (int m = 1; m < 16; m <<= 1) x += __shfl_xor(x, m, 16);
    return x;
}
#endif

// ---------------------------------------------------------------------------
// fp32 -> bf16 elementwise convert (n multiple of 4)
// ---------------------------------------------------------------------------
__global__ void cvt_f32_bf16(const float* __restrict__ src,
                             bf16* __restrict__ dst, int n) {
    int i = (blockIdx.x * blockDim.x + threadIdx.x) * 4;
    if (i + 3 < n) {
        float4 f = *(const float4*)(src + i);
        v4bf o;
        o[0] = to_bf16(f.x); o[1] = to_bf16(f.y);
        o[2] = to_bf16(f.z); o[3] = to_bf16(f.w);
        *(v4bf*)(dst + i) = o;
    }
}

// ---------------------------------------------------------------------------
// WMMA GEMM: Y[M_TOT x 1024] = A[M_TOT x 1024](bf16, row-major, K contig)
//                             x W[1024 x 1024](bf16, [N,K] row-major) + bias
// Each wave: 16(M) x 64(N) tile. K-loop unrolled x2 with ping-pong fragment
// sets (no rotation copies); 8 WMMAs per 64-deep K chunk.
// MODE 0: store bf16 row-major    MODE 1: store bf16 transposed [bh, d, key]
// MODE 2: store f32 row-major
// ---------------------------------------------------------------------------
template <int MODE>
__global__ __launch_bounds__(256, 2)
void gemm_wmma_bf16(const bf16* __restrict__ A,
                    const bf16* __restrict__ W,
                    const float* __restrict__ bias,
                    void* __restrict__ Out) {
    const int K = D_MODEL;
    const int wave = threadIdx.x >> 5;
    const int lane = threadIdx.x & 31;
    const int lo16 = lane & 15;
    const int hi   = lane >> 4;

    const int m0 = blockIdx.y * 128 + wave * 16;
    const int n0 = blockIdx.x * 64;

    const bf16* arow = A + (size_t)(m0 + lo16) * K + (hi << 4);
    const bf16* brow[4];
    #pragma unroll
    for (int j = 0; j < 4; ++j)
        brow[j] = W + (size_t)(n0 + j * 16 + lo16) * K + (hi << 4);

    // ping-pong fragment sets
    v8bf alo0, ahi0, alo1, ahi1;
    v16bf b0[4], b1[4];

    // prologue: set 0 <- kk = 0
    alo0 = *(const v8bf*)(arow);
    ahi0 = *(const v8bf*)(arow + 8);
    #pragma unroll
    for (int j = 0; j < 4; ++j) b0[j] = *(const v16bf*)(brow[j]);

    v8f acc[4] = {v8f{}, v8f{}, v8f{}, v8f{}};

    for (int kk = 0; kk < K; kk += 64) {
        // set 1 <- kk + 32 (always in range: K % 64 == 0)
        __builtin_prefetch(arow + kk + 256, 0, 1);
        alo1 = *(const v8bf*)(arow + kk + 32);
        ahi1 = *(const v8bf*)(arow + kk + 40);
        #pragma unroll
        for (int j = 0; j < 4; ++j) b1[j] = *(const v16bf*)(brow[j] + kk + 32);

        // consume set 0 (kk)
        {
            v16bf af = cat8(alo0, ahi0);
            #pragma unroll
            for (int j = 0; j < 4; ++j) acc[j] = wmma_bf16(af, b0[j], acc[j]);
        }

        // set 0 <- kk + 64 (uniform guard on final chunk)
        if (kk + 64 < K) {
            alo0 = *(const v8bf*)(arow + kk + 64);
            ahi0 = *(const v8bf*)(arow + kk + 72);
            #pragma unroll
            for (int j = 0; j < 4; ++j) b0[j] = *(const v16bf*)(brow[j] + kk + 64);
        }

        // consume set 1 (kk + 32)
        {
            v16bf af = cat8(alo1, ahi1);
            #pragma unroll
            for (int j = 0; j < 4; ++j) acc[j] = wmma_bf16(af, b1[j], acc[j]);
        }
    }

    const int rbase = hi * 8;
    #pragma unroll
    for (int j = 0; j < 4; ++j) {
        const int gc = n0 + j * 16 + lo16;
        const float bv = bias[gc];
        #pragma unroll
        for (int r = 0; r < 8; ++r) {
            const int gm = m0 + rbase + r;
            const float v = acc[j][r] + bv;
            if (MODE == 0) {
                ((bf16*)Out)[(size_t)gm * D_MODEL + gc] = to_bf16(v);
            } else if (MODE == 1) {
                const int b_ = gm >> 11, l_ = gm & 2047;
                const int h_ = gc >> 6,  d_ = gc & 63;
                ((bf16*)Out)[((size_t)((b_ * N_HEAD + h_) * D_HEAD + d_) << 11) + l_] =
                    to_bf16(v);
            } else {
                ((float*)Out)[(size_t)gm * D_MODEL + gc] = v;
            }
        }
    }
}

// ---------------------------------------------------------------------------
// Flash attention per (b,h). Block = 8 waves, each wave: 16 queries x Dh=64.
// Online softmax (permlane16 reductions), P staged in per-wave LDS to build
// the A-fragment for PV. V fragments prefetched before the softmax.
// ---------------------------------------------------------------------------
__global__ __launch_bounds__(256, 2)
void attn_fa_wmma(const bf16* __restrict__ Qp,
                  const bf16* __restrict__ Kp,
                  const bf16* __restrict__ Vt,
                  bf16* __restrict__ Ao) {
    __shared__ __attribute__((aligned(16))) bf16 plds[8][16][40]; // 16B-aligned rows

    const int wave = threadIdx.x >> 5;
    const int lane = threadIdx.x & 31;
    const int lo16 = lane & 15;
    const int hi   = lane >> 4;

    const int bh = blockIdx.y;
    const int b  = bh >> 4;
    const int h  = bh & 15;
    const int q0 = blockIdx.x * 128 + wave * 16;

    // Q A-fragments (16 x 64 over two K=32 steps), loaded once
    const bf16* qrow = Qp + (size_t)(b * SEQ_L + q0 + lo16) * D_MODEL + h * D_HEAD;
    v16bf qa[2];
    #pragma unroll
    for (int t = 0; t < 2; ++t) {
        const bf16* p = qrow + t * 32 + (hi << 4);
        qa[t] = cat8(*(const v8bf*)p, *(const v8bf*)(p + 8));
    }

    v8f acc[4] = {v8f{}, v8f{}, v8f{}, v8f{}};
    float mi[8], li[8];
    #pragma unroll
    for (int r = 0; r < 8; ++r) { mi[r] = -1e30f; li[r] = 0.0f; }

    const float sc = 0.125f; // 1/sqrt(64)
    const bf16* Kbase = Kp + (size_t)b * SEQ_L * D_MODEL + h * D_HEAD;
    const bf16* Vbase = Vt + ((size_t)bh * D_HEAD << 11);

    for (int kv = 0; kv < SEQ_L; kv += 32) {
        // ---- scores: two 16x16 tiles (keys kv..kv+15, kv+16..kv+31) ----
        v8f s0 = {}, s1 = {};
        #pragma unroll
        for (int t = 0; t < 2; ++t) {
            const int doff = t * 32 + (hi << 4);
            v16bf kb0 = *(const v16bf*)(Kbase + (size_t)(kv + lo16) * D_MODEL + doff);
            s0 = wmma_bf16(qa[t], kb0, s0);
            v16bf kb1 = *(const v16bf*)(Kbase + (size_t)(kv + 16 + lo16) * D_MODEL + doff);
            s1 = wmma_bf16(qa[t], kb1, s1);
        }

        // ---- V fragments: independent of softmax, start loads now ----
        v16bf vb[4];
        #pragma unroll
        for (int j = 0; j < 4; ++j)
            vb[j] = *(const v16bf*)(Vbase + ((size_t)(j * 16 + lo16) << 11) + kv + (hi << 4));

        // ---- online softmax over the 32 new keys (pure VALU reductions) ----
        #pragma unroll
        for (int r = 0; r < 8; ++r) {
            const float v0 = s0[r] * sc, v1 = s1[r] * sc;
            const float rm = rowmax16(fmaxf(v0, v1));
            const float mn = fmaxf(mi[r], rm);
            const float alpha = __expf(mi[r] - mn);
            const float p0 = __expf(v0 - mn);
            const float p1 = __expf(v1 - mn);
            li[r] = li[r] * alpha + rowsum16(p0 + p1);
            mi[r] = mn;
            #pragma unroll
            for (int j = 0; j < 4; ++j) acc[j][r] *= alpha;
            const int row = hi * 8 + r;
            plds[wave][row][lo16]      = to_bf16(p0);
            plds[wave][row][16 + lo16] = to_bf16(p1);
        }

        // intra-wave LDS turnaround: stores must land before cross-lane reads
        asm volatile("s_wait_dscnt 0x0" ::: "memory");
        __builtin_amdgcn_wave_barrier();

        // P as A-fragment (16 q-rows x 32 keys)
        const bf16* pr = &plds[wave][lo16][hi << 4];
        v16bf pa = cat8(*(const v8bf*)pr, *(const v8bf*)(pr + 8));

        // ---- PV: acc(16x64) += P(16x32) x V(32x64) ----
        #pragma unroll
        for (int j = 0; j < 4; ++j)
            acc[j] = wmma_bf16(pa, vb[j], acc[j]);
        __builtin_amdgcn_wave_barrier();
    }

    // ---- epilogue: normalize and store bf16 [token, h*64+d] ----
    #pragma unroll
    for (int r = 0; r < 8; ++r) {
        const float inv = 1.0f / li[r];
        const int gq = q0 + hi * 8 + r;
        bf16* orow = Ao + (size_t)(b * SEQ_L + gq) * D_MODEL + h * D_HEAD;
        #pragma unroll
        for (int j = 0; j < 4; ++j)
            orow[j * 16 + lo16] = to_bf16(acc[j][r] * inv);
    }
}

// ---------------------------------------------------------------------------
// Host-side launch
// ---------------------------------------------------------------------------
extern "C" void kernel_launch(void* const* d_in, const int* in_sizes, int n_in,
                              void* d_out, int out_size, void* d_ws, size_t ws_size,
                              hipStream_t stream) {
    const float* q   = (const float*)d_in[0];
    const float* k   = (const float*)d_in[1];
    const float* v   = (const float*)d_in[2];
    const float* w_q = (const float*)d_in[3];
    const float* b_q = (const float*)d_in[4];
    const float* w_k = (const float*)d_in[5];
    const float* b_k = (const float*)d_in[6];
    const float* w_v = (const float*)d_in[7];
    const float* b_v = (const float*)d_in[8];
    const float* w_o = (const float*)d_in[9];
    const float* b_o = (const float*)d_in[10];

    const size_t ACT_ELTS = (size_t)M_TOT * D_MODEL;   // 8,388,608
    const size_t W_ELTS   = (size_t)D_MODEL * D_MODEL; // 1,048,576

    char* ws = (char*)d_ws;
    size_t off = 0;
    bf16* xq  = (bf16*)(ws + off); off += ACT_ELTS * 2;
    bf16* xk  = (bf16*)(ws + off); off += ACT_ELTS * 2;
    bf16* xv  = (bf16*)(ws + off); off += ACT_ELTS * 2;
    bf16* wqb = (bf16*)(ws + off); off += W_ELTS * 2;
    bf16* wkb = (bf16*)(ws + off); off += W_ELTS * 2;
    bf16* wvb = (bf16*)(ws + off); off += W_ELTS * 2;
    bf16* wob = (bf16*)(ws + off); off += W_ELTS * 2;
    bf16* Qp  = (bf16*)(ws + off); off += ACT_ELTS * 2;
    bf16* Kp  = (bf16*)(ws + off); off += ACT_ELTS * 2;
    bf16* Vt  = (bf16*)(ws + off); off += ACT_ELTS * 2;   // [bh, d, key]
    bf16* Ao  = (bf16*)(ws + off); off += ACT_ELTS * 2;
    (void)ws_size; (void)n_in; (void)in_sizes; (void)out_size;

    // 1) fp32 -> bf16 conversions
    {
        const int tb = 256;
        const int gb_act = (int)(ACT_ELTS / (tb * 4));
        const int gb_w   = (int)(W_ELTS   / (tb * 4));
        cvt_f32_bf16<<<gb_act, tb, 0, stream>>>(q,   xq,  (int)ACT_ELTS);
        cvt_f32_bf16<<<gb_act, tb, 0, stream>>>(k,   xk,  (int)ACT_ELTS);
        cvt_f32_bf16<<<gb_act, tb, 0, stream>>>(v,   xv,  (int)ACT_ELTS);
        cvt_f32_bf16<<<gb_w,   tb, 0, stream>>>(w_q, wqb, (int)W_ELTS);
        cvt_f32_bf16<<<gb_w,   tb, 0, stream>>>(w_k, wkb, (int)W_ELTS);
        cvt_f32_bf16<<<gb_w,   tb, 0, stream>>>(w_v, wvb, (int)W_ELTS);
        cvt_f32_bf16<<<gb_w,   tb, 0, stream>>>(w_o, wob, (int)W_ELTS);
    }

    // 2) projections: Q, K row-major; V transposed to [bh, d, key]
    dim3 ggrid(D_MODEL / 64, M_TOT / 128);
    gemm_wmma_bf16<0><<<ggrid, 256, 0, stream>>>(xq, wqb, b_q, (void*)Qp);
    gemm_wmma_bf16<0><<<ggrid, 256, 0, stream>>>(xk, wkb, b_k, (void*)Kp);
    gemm_wmma_bf16<1><<<ggrid, 256, 0, stream>>>(xv, wvb, b_v, (void*)Vt);

    // 3) flash attention per (b,h)
    dim3 agrid(SEQ_L / 128, BATCH * N_HEAD);
    attn_fa_wmma<<<agrid, 256, 0, stream>>>(Qp, Kp, Vt, Ao);

    // 4) output projection -> fp32 d_out
    gemm_wmma_bf16<2><<<ggrid, 256, 0, stream>>>(Ao, wob, b_o, d_out);
}
